// DistanceContainedConv3d_6682969113119
// MI455X (gfx1250) — matrix operand order repaired and checked by reading, not compiled, not run on hardware
//
#include <hip/hip_runtime.h>
#include <hip/hip_bf16.h>
#include <math.h>

typedef __attribute__((ext_vector_type(16))) _Float16 v16h;
typedef __attribute__((ext_vector_type(8)))  _Float16 v8h;
typedef __attribute__((ext_vector_type(8)))  float    v8f;

#define NPTS 16384
#define KNN  32
#define CI   64
#define CO   64
#define BDIM 27
#define BPAD 32
#define KKD  (BPAD * CI)   // 2048 padded contraction dim for gemm2
#define CHUNK 256          // K-halves staged per async chunk (64 rows x 512B = 32KB)
#define NCHUNK (KKD / CHUNK)

static __device__ inline v16h cat16(v8h lo, v8h hi) {
  v16h r;
#pragma unroll
  for (int e = 0; e < 8; ++e) { r[e] = lo[e]; r[e + 8] = hi[e]; }
  return r;
}

// ---------------------------------------------------------------------------
// Kernel 1: exact KNN (K=32) over all 16384 points.
// One thread per query, LDS-tiled candidates, register-resident sorted top-32.
// ---------------------------------------------------------------------------
__global__ __launch_bounds__(256) void knn_kernel(const float* __restrict__ pos,
                                                  int* __restrict__ out_idx) {
  int q = blockIdx.x * 256 + threadIdx.x;
  float px = pos[q * 3 + 0], py = pos[q * 3 + 1], pz = pos[q * 3 + 2];
  float sqp = px * px + py * py + pz * pz;
  float dl[KNN]; int il[KNN];
#pragma unroll
  for (int j = 0; j < KNN; ++j) { dl[j] = 3.4e38f; il[j] = 0; }
  __shared__ float4 tile[256];
  for (int t = 0; t < NPTS / 256; ++t) {
    __syncthreads();
    int c = t * 256 + threadIdx.x;
    float cx = pos[c * 3], cy = pos[c * 3 + 1], cz = pos[c * 3 + 2];
    tile[threadIdx.x] = make_float4(cx, cy, cz, cx * cx + cy * cy + cz * cz);
    __syncthreads();
    for (int j = 0; j < 256; ++j) {
      float4 o = tile[j];
      float d = sqp + o.w - 2.0f * (px * o.x + py * o.y + pz * o.z);
      int ci = t * 256 + j;
      if (d < dl[KNN - 1]) {
        float cv = d; int cc = ci;
#pragma unroll
        for (int s = 0; s < KNN; ++s) {
          if (cv < dl[s]) {
            float tv = dl[s]; dl[s] = cv; cv = tv;
            int   ti = il[s]; il[s] = cc; cc = ti;
          }
        }
      }
    }
  }
#pragma unroll
  for (int j = 0; j < KNN; ++j) out_idx[q * KNN + j] = il[j];
}

// ---------------------------------------------------------------------------
// Kernel 2: per-point PCA (3x3 Jacobi eigh) + spherical coords + poly basis.
// Writes basis TRANSPOSED, f16, padded:  basisT[n][b (32)][k (32)]
// ---------------------------------------------------------------------------
static __device__ inline void jrot(float A[3][3], float V[3][3], int p, int qq) {
  float apq = A[p][qq];
  if (fabsf(apq) > 1e-12f) {
    float th = (A[qq][qq] - A[p][p]) / (2.0f * apq);
    float t  = copysignf(1.0f, th) / (fabsf(th) + sqrtf(th * th + 1.0f));
    float c  = rsqrtf(t * t + 1.0f);
    float s  = t * c;
#pragma unroll
    for (int k = 0; k < 3; ++k) { float a0 = A[k][p], a1 = A[k][qq]; A[k][p] = c * a0 - s * a1; A[k][qq] = s * a0 + c * a1; }
#pragma unroll
    for (int k = 0; k < 3; ++k) { float a0 = A[p][k], a1 = A[qq][k]; A[p][k] = c * a0 - s * a1; A[qq][k] = s * a0 + c * a1; }
#pragma unroll
    for (int k = 0; k < 3; ++k) { float v0 = V[k][p], v1 = V[k][qq]; V[k][p] = c * v0 - s * v1; V[k][qq] = s * v0 + c * v1; }
  }
}

static __device__ inline void cswap(float ev[3], float V[3][3], int i, int j) {
  if (ev[i] < ev[j]) {
    float t = ev[i]; ev[i] = ev[j]; ev[j] = t;
#pragma unroll
    for (int r = 0; r < 3; ++r) { float tv = V[r][i]; V[r][i] = V[r][j]; V[r][j] = tv; }
  }
}

__global__ __launch_bounds__(256) void geom_kernel(const float* __restrict__ pos,
                                                   const int* __restrict__ idx,
                                                   _Float16* __restrict__ basisT) {
  int q = blockIdx.x * 256 + threadIdx.x;
  const int* nb = idx + q * KNN;
  const float invK = 1.0f / (float)KNN;
  float mx = 0.f, my = 0.f, mz = 0.f;
  for (int k = 0; k < KNN; ++k) {
    int j = nb[k];
    mx += pos[j * 3]; my += pos[j * 3 + 1]; mz += pos[j * 3 + 2];
  }
  mx *= invK; my *= invK; mz *= invK;
  float A[3][3] = {{0.f,0.f,0.f},{0.f,0.f,0.f},{0.f,0.f,0.f}};
  for (int k = 0; k < KNN; ++k) {
    int j = nb[k];
    float dx = pos[j * 3] - mx, dy = pos[j * 3 + 1] - my, dz = pos[j * 3 + 2] - mz;
    A[0][0] += dx * dx; A[0][1] += dx * dy; A[0][2] += dx * dz;
    A[1][1] += dy * dy; A[1][2] += dy * dz; A[2][2] += dz * dz;
  }
  A[0][0] *= invK; A[0][1] *= invK; A[0][2] *= invK;
  A[1][1] *= invK; A[1][2] *= invK; A[2][2] *= invK;
  A[1][0] = A[0][1]; A[2][0] = A[0][2]; A[2][1] = A[1][2];
  float V[3][3] = {{1.f,0.f,0.f},{0.f,1.f,0.f},{0.f,0.f,1.f}};
#pragma unroll
  for (int sw = 0; sw < 6; ++sw) { jrot(A, V, 0, 1); jrot(A, V, 0, 2); jrot(A, V, 1, 2); }
  float ev[3] = {A[0][0], A[1][1], A[2][2]};
  cswap(ev, V, 0, 1); cswap(ev, V, 0, 2); cswap(ev, V, 1, 2);   // descending

  _Float16* bt = basisT + (size_t)q * (BPAD * KNN);
  for (int k = 0; k < KNN; ++k) {
    int j = nb[k];
    float dx = pos[j * 3] - mx, dy = pos[j * 3 + 1] - my, dz = pos[j * 3 + 2] - mz;
    float x = dx * V[0][0] + dy * V[1][0] + dz * V[2][0];
    float y = dx * V[0][1] + dy * V[1][1] + dz * V[2][1];
    float z = dx * V[0][2] + dy * V[1][2] + dz * V[2][2];
    float r = sqrtf(x * x + y * y + z * z + 1e-8f);
    float cz = fminf(fmaxf(z / r, -1.0f + 1e-6f), 1.0f - 1e-6f);
    float th = acosf(cz);
    float ph = atan2f(y, x);
    float pr[3] = {1.0f, r, r * r};
    float cb[3] = {1.0f, cosf(th), cosf(2.0f * th)};
    float cc[3] = {1.0f, cosf(ph), cosf(2.0f * ph)};
#pragma unroll
    for (int a = 0; a < 3; ++a)
#pragma unroll
      for (int b = 0; b < 3; ++b)
#pragma unroll
        for (int c = 0; c < 3; ++c)
          bt[(a * 9 + b * 3 + c) * KNN + k] = (_Float16)(pr[a] * cb[b] * cc[c]);
  }
  for (int b = BDIM; b < BPAD; ++b)
    for (int k = 0; k < KNN; ++k) bt[b * KNN + k] = (_Float16)0.0f;
}

// ---------------------------------------------------------------------------
// Kernel 3: pack coeff (27,64,64) f32 -> Ct[o][b*64+i] f16, zero-padded b>=27.
// ---------------------------------------------------------------------------
__global__ __launch_bounds__(256) void pack_kernel(const float* __restrict__ coeff,
                                                   _Float16* __restrict__ Ct) {
  int tid = blockIdx.x * 256 + threadIdx.x;   // 0 .. CO*KKD-1
  int o  = tid >> 11;                          // / 2048
  int kk = tid & (KKD - 1);
  int b = kk >> 6, i = kk & 63;
  float v = (b < BDIM) ? coeff[b * CI * CO + i * CO + o] : 0.0f;
  Ct[(size_t)o * KKD + kk] = (_Float16)v;
}

// ---------------------------------------------------------------------------
// Kernel 4: per-point G = basis^T (32x32) * gathered feats (32x64), WMMA f16.
// ---------------------------------------------------------------------------
__global__ __launch_bounds__(256) void gemm1_kernel(const float* __restrict__ chan,
                                                    const int* __restrict__ idx,
                                                    const _Float16* __restrict__ basisT,
                                                    _Float16* __restrict__ G) {
  __shared__ _Float16 fT[8][CI * KNN];   // per-wave feats transposed [i][k]
  int lane = threadIdx.x & 31, wave = threadIdx.x >> 5;
  int n = blockIdx.x * 8 + wave;
  int nbr = idx[n * KNN + lane];
  const float* row = chan + (size_t)nbr * CI;
#pragma unroll
  for (int i = 0; i < CI; ++i) fT[wave][i * KNN + lane] = (_Float16)row[i];
  __syncthreads();

  int m  = lane & 15;
  int kb = (lane >> 4) * 8;
  const _Float16* bb = basisT + (size_t)n * (BPAD * KNN);
  v16h a0 = cat16(*(const v8h*)(bb + m * KNN + kb),        *(const v8h*)(bb + m * KNN + kb + 16));
  v16h a1 = cat16(*(const v8h*)(bb + (16 + m) * KNN + kb), *(const v8h*)(bb + (16 + m) * KNN + kb + 16));

  v8f zero = {};
  v8f acc[8];
#pragma unroll
  for (int j = 0; j < 8; ++j) acc[j] = zero;

  int koff = (lane >> 4) * 8;
#pragma unroll
  for (int nt = 0; nt < 4; ++nt) {
    int icol = nt * 16 + m;
    const _Float16* bp = &fT[wave][icol * KNN + koff];
    v16h bfrag = cat16(*(const v8h*)bp, *(const v8h*)(bp + 16));
    acc[2 * nt + 0] = __builtin_amdgcn_wmma_f32_16x16x32_f16(false, a0, false, bfrag, (short)0, acc[2 * nt + 0], false, false);
    acc[2 * nt + 1] = __builtin_amdgcn_wmma_f32_16x16x32_f16(false, a1, false, bfrag, (short)0, acc[2 * nt + 1], false, false);
  }

  _Float16* g = G + (size_t)n * KKD;
#pragma unroll
  for (int nt = 0; nt < 4; ++nt) {
    int icol = nt * 16 + m;
#pragma unroll
    for (int mt = 0; mt < 2; ++mt) {
      v8f c = acc[2 * nt + mt];
#pragma unroll
      for (int r = 0; r < 8; ++r) {
        int mr = mt * 16 + ((lane >> 4) ? 8 + r : r);
        g[mr * CI + icol] = (_Float16)c[r];
      }
    }
  }
}

// ---------------------------------------------------------------------------
// Kernel 5: out[n][o] = (1/K) * G[n][kk] x Ct[kk][o], kk=2048, WMMA f16.
// Coeff panel staged into LDS in 32KB chunks with gfx1250 async-to-LDS DMA
// (global_load_async_to_lds_b128, ASYNCcnt), double-buffered and overlapped
// with WMMA compute. B-fragments are then ds_load_b128 from LDS.
// ---------------------------------------------------------------------------
__global__ __launch_bounds__(256) void gemm2_kernel(const _Float16* __restrict__ G,
                                                    const _Float16* __restrict__ Ct,
                                                    float* __restrict__ out) {
  __shared__ _Float16 bbuf[2][CO * CHUNK];   // 2 x 32KB double buffer
  int tid  = threadIdx.x;
  int lane = tid & 31, wave = tid >> 5;
  int n0 = blockIdx.x * 128 + wave * 16;

  int m    = lane & 15;
  int kb8  = (lane >> 4) * 8;    // A-frag half-K base
  int kb16 = (lane >> 4) * 8;    // B-frag half-K base
  const _Float16* grow = G + (size_t)(n0 + m) * KKD;
  __builtin_prefetch(grow, 0, 0);            // global_prefetch_b8 warms A rows

  // LDS byte offset = low 32 bits of the flat shared-memory address
  // (ISA: LDS aperture occupies addr[63:32]).
  uint32_t lds0 = (uint32_t)(uintptr_t)(&bbuf[0][0]);

  // Cooperative async stage of one 64 x CHUNK chunk of Ct into LDS:
  // 2048 b128 transfers per chunk, 8 per thread.
  auto stage = [&](int chunk, int buf) {
#pragma unroll
    for (int i = 0; i < 8; ++i) {
      int e  = tid + i * 256;                       // b128 unit index, 0..2047
      int o  = e >> 5;                              // coeff row (32 x b128 per row)
      int cb = (e & 31) * 16;                       // byte within row chunk
      const char* gsrc = (const char*)Ct + (size_t)o * (KKD * 2) + (size_t)chunk * (CHUNK * 2) + cb;
      uint32_t ldst = lds0 + (uint32_t)buf * (CO * CHUNK * 2) + (uint32_t)e * 16;
      asm volatile("global_load_async_to_lds_b128 %0, %1, off"
                   :: "v"(ldst), "v"(gsrc) : "memory");
    }
  };

  stage(0, 0);

  v8f zero = {};
  v8f acc[4];
#pragma unroll
  for (int j = 0; j < 4; ++j) acc[j] = zero;

  for (int c = 0; c < NCHUNK; ++c) {
    if (c + 1 < NCHUNK) {
      stage(c + 1, (c + 1) & 1);
      asm volatile("s_wait_asynccnt 0x8" ::: "memory");   // chunk c's 8 ops done
    } else {
      asm volatile("s_wait_asynccnt 0x0" ::: "memory");
    }
    __syncthreads();                                      // chunk c visible to all waves

    const _Float16* bbase = &bbuf[c & 1][0];
#pragma unroll
    for (int ksl = 0; ksl < CHUNK / 32; ++ksl) {
      int ks = c * (CHUNK / 32) + ksl;
      const _Float16* ap = grow + ks * 32 + kb8;
      v16h a = cat16(*(const v8h*)ap, *(const v8h*)(ap + 16));
#pragma unroll
      for (int nt = 0; nt < 4; ++nt) {
        const _Float16* bp = bbase + (nt * 16 + m) * CHUNK + ksl * 32 + kb16;
        v16h b = cat16(*(const v8h*)bp, *(const v8h*)(bp + 16));
        acc[nt] = __builtin_amdgcn_wmma_f32_16x16x32_f16(false, a, false, b, (short)0, acc[nt], false, false);
      }
    }
    __syncthreads();   // protect double buffer before next overwrite
  }

  const float inv = 1.0f / (float)KNN;
#pragma unroll
  for (int nt = 0; nt < 4; ++nt) {
    int o = nt * 16 + m;
#pragma unroll
    for (int r = 0; r < 8; ++r) {
      int n = n0 + ((lane >> 4) ? 8 + r : r);
      out[(size_t)o * NPTS + n] = acc[nt][r] * inv;
    }
  }
}

// ---------------------------------------------------------------------------
extern "C" void kernel_launch(void* const* d_in, const int* in_sizes, int n_in,
                              void* d_out, int out_size, void* d_ws, size_t ws_size,
                              hipStream_t stream) {
  (void)in_sizes; (void)n_in; (void)out_size; (void)ws_size;
  const float* pos   = (const float*)d_in[0];   // (16384, 3)
  const float* chan  = (const float*)d_in[1];   // (16384, 64)
  const float* coeff = (const float*)d_in[2];   // (27, 64, 64)

  char* ws = (char*)d_ws;
  size_t off = 0;
  auto carve = [&](size_t bytes) -> void* {
    void* p = ws + off;
    off = (off + bytes + 255) & ~(size_t)255;
    return p;
  };
  int*      d_idx   = (int*)     carve((size_t)NPTS * KNN * sizeof(int));   //   2 MB
  _Float16* d_basis = (_Float16*)carve((size_t)NPTS * BPAD * KNN * 2);      //  32 MB
  _Float16* d_Ct    = (_Float16*)carve((size_t)CO * KKD * 2);               // 256 KB
  _Float16* d_G     = (_Float16*)carve((size_t)NPTS * KKD * 2);             //  64 MB

  knn_kernel  <<<NPTS / 256, 256, 0, stream>>>(pos, d_idx);
  geom_kernel <<<NPTS / 256, 256, 0, stream>>>(pos, d_idx, d_basis);
  pack_kernel <<<(CO * KKD) / 256, 256, 0, stream>>>(coeff, d_Ct);
  gemm1_kernel<<<NPTS / 8,   256, 0, stream>>>(chan, d_idx, d_basis, d_G);
  gemm2_kernel<<<NPTS / 128, 256, 0, stream>>>(d_G, d_Ct, (float*)d_out);
}
